// GGNNModel_29472065585398
// MI455X (gfx1250) — matrix operand order repaired and checked by reading, not compiled, not run on hardware
//
#include <hip/hip_runtime.h>
#include <hip/hip_bf16.h>
#include <stdint.h>

// ---------------------------------------------------------------------------
// CDNA5 (gfx1250) gated-GCN forward.  All GEMMs run on v_wmma_f32_16x16x32_bf16.
// WMMA operands are kept in LDS in *fragment-major* layout so every fragment
// load is two ds_load_b128 directly into the contiguous even-aligned VGPRs.
// ---------------------------------------------------------------------------

typedef __attribute__((ext_vector_type(16))) __bf16        v16bf;
typedef __attribute__((ext_vector_type(8)))  float         v8f;
typedef __attribute__((ext_vector_type(8)))  unsigned int  v8u;

#define LN_EPS 1e-5f

__device__ __forceinline__ unsigned short f2bf(float f) {
  unsigned int u = __builtin_bit_cast(unsigned int, f);
  u += 0x7FFFu + ((u >> 16) & 1u);            // round to nearest even
  return (unsigned short)(u >> 16);
}
__device__ __forceinline__ float reluf(float x)    { return fmaxf(x, 0.f); }
__device__ __forceinline__ float eluf(float x)     { return x > 0.f ? x : (__expf(x) - 1.f); }
__device__ __forceinline__ float sigmoidf_(float x){ return 1.f / (1.f + __expf(-x)); }

__device__ __forceinline__ void store_f4(float* p, float4 v) {
  p[0] = v.x; p[1] = v.y; p[2] = v.z; p[3] = v.w;
}

// ---- fragment-major LDS layout helpers ------------------------------------
// A-matrix (16xK, bf16), per ISA 7.12.2:
//   lane = m + 16*hb with hb = (k>>3)&1 ;  r = 4*(klocal>>4) + ((k&7)>>1)
//   fragment of lane for k-slice ks is 16 contiguous bf16 (32 B).
__device__ __forceinline__ int afrag_idx(int m, int k) {
  int ks = k >> 5, kl = k & 31;
  int hb = (kl >> 3) & 1;
  int r  = ((kl >> 4) << 2) + ((kl & 7) >> 1);
  return (((ks << 5) + (hb << 4) + m) << 4) + (r << 1) + (kl & 1);
}
// B-matrix (Kx16n, bf16): lane = (n&15) + 16*((klocal>>3)&1); elem = 8*(klocal>>4) + (klocal&7)
__device__ __forceinline__ int bfrag_idx(int k, int n, int NT) {
  int ks = k >> 5, kl = k & 31;
  int lane = (n & 15) + (((kl >> 3) & 1) << 4);
  int nt = n >> 4;
  return (((ks * NT + nt) << 5) + lane) * 16 + ((kl >> 4) << 3) + (kl & 7);
}

__device__ __forceinline__ v16bf lds_load_frag(const unsigned short* p) {
  return __builtin_bit_cast(v16bf, *(const v8u*)p);      // 2x ds_load_b128
}
__device__ __forceinline__ const unsigned short* afrag_ptr(const unsigned short* base, int ks, int lane) {
  return base + (((ks << 5) + lane) << 4);
}
__device__ __forceinline__ const unsigned short* bfrag_ptr(const unsigned short* base, int NT, int ks, int nt, int lane) {
  return base + (((ks * NT + nt) << 5) + lane) * 16;
}

// store 4 consecutive k values (c % 4 == 0) of row m: contiguous in frag layout
__device__ __forceinline__ void store_bf4_frag(unsigned short* base, int m, int c, float4 v) {
  uint2 u;
  u.x = (unsigned)f2bf(v.x) | ((unsigned)f2bf(v.y) << 16);
  u.y = (unsigned)f2bf(v.z) | ((unsigned)f2bf(v.w) << 16);
  *(uint2*)(base + afrag_idx(m, c)) = u;                 // ds_store_b64
}

__device__ __forceinline__ v8f wmma_bf16(v16bf a, v16bf b, v8f c) {
  return __builtin_amdgcn_wmma_f32_16x16x32_bf16(false, a, false, b, (short)0, c, false, false);
}

// Stage global row-major W[K][NCOLS] (f32) into LDS bf16, fragment-major.
template <int K, int NCOLS>
__device__ __forceinline__ void stage_w_frag(const float* __restrict__ W,
                                             unsigned short* dst, int tid, int nthreads) {
  constexpr int NT = NCOLS / 16;
  for (int i = tid; i < K * NCOLS; i += nthreads) {
    int k = i / NCOLS, n = i % NCOLS;
    dst[bfrag_idx(k, n, NT)] = f2bf(W[i]);
  }
}

// ---------------------------------------------------------------------------
// Degree / norm
// ---------------------------------------------------------------------------
__global__ void fill_kernel(float* __restrict__ p, float v, int n) {
  int i = blockIdx.x * 256 + threadIdx.x;
  if (i < n) p[i] = v;
}
__global__ void deg_accum_kernel(const long long* __restrict__ dst, int E, float* __restrict__ deg) {
  int i = blockIdx.x * 256 + threadIdx.x;
  if (i < E) atomicAdd(&deg[(int)dst[i]], 1.0f);
}
__global__ void rsqrt_kernel(float* __restrict__ d, int N) {
  int i = blockIdx.x * 256 + threadIdx.x;
  if (i < N) d[i] = rsqrtf(d[i]);   // deg >= 1 always (self loop)
}
__global__ void norm_kernel(const long long* __restrict__ src, const long long* __restrict__ dst,
                            int Eorig, int Etot, const float* __restrict__ dis, float* __restrict__ nrm) {
  int e = blockIdx.x * 256 + threadIdx.x;
  if (e < Etot) {
    int s = (e < Eorig) ? (int)src[e] : (e - Eorig);
    int d = (e < Eorig) ? (int)dst[e] : (e - Eorig);
    nrm[e] = dis[s] * dis[d];
  }
}

// ---------------------------------------------------------------------------
// Preproc: h0 = relu(LN(x@Wpre+b)); node_id = relu(h0@Wid+b); h = relu(h0@Wh0+b)
// ---------------------------------------------------------------------------
__global__ __launch_bounds__(256) void preproc_kernel(
    const float* __restrict__ x,
    const float* __restrict__ Wpre, const float* __restrict__ bpre,
    const float* __restrict__ lng,  const float* __restrict__ lnb,
    const float* __restrict__ Wid,  const float* __restrict__ bid,
    const float* __restrict__ Wh0,  const float* __restrict__ bh0,
    float* __restrict__ node_id, float* __restrict__ h, int N) {
  __shared__ float sWpre[16 * 32];
  __shared__ float sWid[32 * 64];
  __shared__ float sWh0[32 * 64];
  __shared__ float sB[32 * 3 + 64 * 2];
  int tid = threadIdx.x;
  for (int i = tid; i < 16 * 32; i += 256) sWpre[i] = Wpre[i];
  for (int i = tid; i < 32 * 64; i += 256) { sWid[i] = Wid[i]; sWh0[i] = Wh0[i]; }
  if (tid < 32) { sB[tid] = bpre[tid]; sB[32 + tid] = lng[tid]; sB[64 + tid] = lnb[tid]; }
  if (tid < 64) { sB[96 + tid] = bid[tid]; sB[160 + tid] = bh0[tid]; }
  __syncthreads();
  int node = blockIdx.x * 256 + tid;
  if (node >= N) return;
  float xin[16];
#pragma unroll
  for (int i = 0; i < 16; ++i) xin[i] = x[node * 16 + i];
  float pre[32];
  float mean = 0.f;
  for (int c = 0; c < 32; ++c) {
    float a = sB[c];
#pragma unroll
    for (int k = 0; k < 16; ++k) a += xin[k] * sWpre[k * 32 + c];
    pre[c] = a; mean += a;
  }
  mean *= (1.f / 32.f);
  float var = 0.f;
#pragma unroll
  for (int c = 0; c < 32; ++c) { float d = pre[c] - mean; var += d * d; }
  var *= (1.f / 32.f);
  float rs = rsqrtf(var + LN_EPS);
  float h0v[32];
#pragma unroll
  for (int c = 0; c < 32; ++c) h0v[c] = reluf((pre[c] - mean) * rs * sB[32 + c] + sB[64 + c]);
  for (int c = 0; c < 64; ++c) {
    float ai = sB[96 + c], ah = sB[160 + c];
    for (int k = 0; k < 32; ++k) { float v = h0v[k]; ai += v * sWid[k * 64 + c]; ah += v * sWh0[k * 64 + c]; }
    node_id[node * 64 + c] = reluf(ai);
    h[node * 64 + c]       = reluf(ah);
  }
}

// ---------------------------------------------------------------------------
// Edge kernel: per 16-edge tile per wave
//   A = bf16([h[dst] | h[src]]) (16x128), G1 = elu(A@Wg0[:128]+bg0) (16x32)
//   G2 = elu(G1@Wg1+bg1),  gate = sigmoid(G2@Wgo+bgo)*norm
//   aggr[dst] += gate * h[src]  (global f32 atomics, L2 resident)
// ---------------------------------------------------------------------------
#define EK_WAVES 4
__global__ __launch_bounds__(128) void edge_gate_kernel(
    const float* __restrict__ h, const float* __restrict__ nrm,
    const long long* __restrict__ esrc, const long long* __restrict__ edst,
    int Eorig, int Etot,
    const float* __restrict__ Wg0, const float* __restrict__ bg0,
    const float* __restrict__ Wg1, const float* __restrict__ bg1,
    const float* __restrict__ Wgo, const float* __restrict__ bgo,
    float* __restrict__ aggr) {
  __shared__ alignas(32) unsigned short sWg0p[128 * 32];
  __shared__ alignas(32) unsigned short sWg1p[32 * 32];
  __shared__ float          sSB[32 * 3 + 1];
  __shared__ float          sXj[EK_WAVES][16 * 64];
  __shared__ alignas(32) unsigned short sA[EK_WAVES][16 * 128];
  __shared__ alignas(32) unsigned short sG[EK_WAVES][16 * 32];
  __shared__ float          sG2[EK_WAVES][16 * 32];
  __shared__ float          sGate[EK_WAVES][16];
  __shared__ int            sDst[EK_WAVES][16];

  const int tid = threadIdx.x, lane = tid & 31, wv = tid >> 5;
  stage_w_frag<128, 32>(Wg0, sWg0p, tid, 128);     // row 128 (zero col) unused
  stage_w_frag<32, 32>(Wg1, sWg1p, tid, 128);
  if (tid < 32) { sSB[tid] = bg0[tid]; sSB[32 + tid] = bg1[tid]; sSB[64 + tid] = Wgo[tid]; }
  if (tid == 0) sSB[96] = bgo[0];

  const int base = (blockIdx.x * EK_WAVES + wv) * 16;
  const int m = lane & 15, hf = lane >> 4;
  int e  = base + m;
  int ec = (e < Etot) ? e : (Etot - 1);
  int sj = (ec < Eorig) ? (int)esrc[ec] : (ec - Eorig);
  int di = (ec < Eorig) ? (int)edst[ec] : (ec - Eorig);
  if (hf == 0) sDst[wv][m] = di;
  const float4* Hi = (const float4*)(h + (size_t)di * 64);
  const float4* Hj = (const float4*)(h + (size_t)sj * 64);
#pragma unroll
  for (int q = 0; q < 8; ++q) {
    int f = hf * 8 + q, c = f * 4;
    store_bf4_frag(sA[wv], m, c, Hi[f]);           // x_i -> A cols 0..63
    float4 vj = Hj[f];
    store_f4(&sXj[wv][m * 64 + c], vj);
    store_bf4_frag(sA[wv], m, 64 + c, vj);         // x_j -> A cols 64..127
  }
  __syncthreads();

  // G1 = elu(A @ Wg0 + bg0)   (2 ntiles x 4 ksteps = 8 WMMA)
#pragma unroll
  for (int nt = 0; nt < 2; ++nt) {
    v8f acc;
    float bv = sSB[(lane & 15) + nt * 16];
#pragma unroll
    for (int v = 0; v < 8; ++v) acc[v] = bv;
#pragma unroll
    for (int ks = 0; ks < 4; ++ks) {
      v16bf a = lds_load_frag(afrag_ptr(sA[wv], ks, lane));
      v16bf b = lds_load_frag(bfrag_ptr(sWg0p, 2, ks, nt, lane));
      acc = wmma_bf16(a, b, acc);
    }
    int n = (lane & 15) + nt * 16;
#pragma unroll
    for (int v = 0; v < 8; ++v) sG[wv][afrag_idx(v + hf * 8, n)] = f2bf(eluf(acc[v]));
  }
  __syncthreads();

  // G2 = elu(G1 @ Wg1 + bg1)  (2 WMMA)
#pragma unroll
  for (int nt = 0; nt < 2; ++nt) {
    v8f acc;
    float bv = sSB[32 + (lane & 15) + nt * 16];
#pragma unroll
    for (int v = 0; v < 8; ++v) acc[v] = bv;
    v16bf a = lds_load_frag(afrag_ptr(sG[wv], 0, lane));
    v16bf b = lds_load_frag(bfrag_ptr(sWg1p, 2, 0, nt, lane));
    acc = wmma_bf16(a, b, acc);
    int n = (lane & 15) + nt * 16;
#pragma unroll
    for (int v = 0; v < 8; ++v) sG2[wv][(v + hf * 8) * 32 + n] = eluf(acc[v]);
  }
  __syncthreads();

  if (lane < 16) {
    float dot = sSB[96];
#pragma unroll
    for (int k = 0; k < 32; ++k) dot += sG2[wv][lane * 32 + k] * sSB[64 + k];
    float nv = ((base + lane) < Etot) ? nrm[base + lane] : 0.f;
    sGate[wv][lane] = sigmoidf_(dot) * nv;
  }
  __syncthreads();

  if (e < Etot) {
    float coef = sGate[wv][m];
    float* ap = aggr + (size_t)sDst[wv][m] * 64;
#pragma unroll
    for (int q = 0; q < 8; ++q) {
      int c = hf * 32 + q * 4;
      atomicAdd(&ap[c + 0], coef * sXj[wv][m * 64 + c + 0]);
      atomicAdd(&ap[c + 1], coef * sXj[wv][m * 64 + c + 1]);
      atomicAdd(&ap[c + 2], coef * sXj[wv][m * 64 + c + 2]);
      atomicAdd(&ap[c + 3], coef * sXj[wv][m * 64 + c + 3]);
    }
  }
}

// ---------------------------------------------------------------------------
// GRU + skip-dense per 16-node tile per wave.
// Two weight phases in LDS (Wz,Wr then Wh,Wd) to stay < 64KB.
// ---------------------------------------------------------------------------
#define GK_WAVES 2
__global__ __launch_bounds__(64) void gru_kernel(
    float* __restrict__ h, const float* __restrict__ aggr, const float* __restrict__ node_id,
    const float* __restrict__ Wz, const float* __restrict__ bz,
    const float* __restrict__ Wr, const float* __restrict__ br,
    const float* __restrict__ Wh, const float* __restrict__ bh,
    const float* __restrict__ Wd, const float* __restrict__ bd, int N) {
  __shared__ alignas(32) unsigned short sWa[128 * 64];
  __shared__ alignas(32) unsigned short sWb[128 * 64];
  __shared__ float          sB[64 * 4];
  __shared__ float          sH[GK_WAVES][16 * 64];
  __shared__ alignas(32) unsigned short sA[GK_WAVES][16 * 128];

  const int tid = threadIdx.x, lane = tid & 31, wv = tid >> 5;
  stage_w_frag<128, 64>(Wz, sWa, tid, 64);
  stage_w_frag<128, 64>(Wr, sWb, tid, 64);
  if (tid < 64) { sB[tid] = bz[tid]; sB[64 + tid] = br[tid]; sB[128 + tid] = bh[tid]; sB[192 + tid] = bd[tid]; }

  const int base = (blockIdx.x * GK_WAVES + wv) * 16;
  const int m = lane & 15, hf = lane >> 4, n0 = lane & 15;
  int node = base + m; if (node >= N) node = N - 1;
  const float4* Hr = (const float4*)(h + (size_t)node * 64);
  const float4* Ar = (const float4*)(aggr + (size_t)node * 64);
#pragma unroll
  for (int q = 0; q < 8; ++q) {
    int f = hf * 8 + q, c = f * 4;
    float4 hv = Hr[f];
    store_f4(&sH[wv][m * 64 + c], hv);
    store_bf4_frag(sA[wv], m, c, hv);
    store_bf4_frag(sA[wv], m, 64 + c, Ar[f]);
  }
  __syncthreads();

  // z, r = sigmoid([h|aggr] @ {Wz,Wr} + b)   (2 x 16 WMMA)
  v8f zf[4], rf[4];
#pragma unroll
  for (int nt = 0; nt < 4; ++nt) {
    v8f az, ar;
    float bvz = sB[n0 + nt * 16], bvr = sB[64 + n0 + nt * 16];
#pragma unroll
    for (int v = 0; v < 8; ++v) { az[v] = bvz; ar[v] = bvr; }
#pragma unroll
    for (int ks = 0; ks < 4; ++ks) {
      v16bf a  = lds_load_frag(afrag_ptr(sA[wv], ks, lane));
      v16bf b1 = lds_load_frag(bfrag_ptr(sWa, 4, ks, nt, lane));
      v16bf b2 = lds_load_frag(bfrag_ptr(sWb, 4, ks, nt, lane));
      az = wmma_bf16(a, b1, az);
      ar = wmma_bf16(a, b2, ar);
    }
#pragma unroll
    for (int v = 0; v < 8; ++v) { zf[nt][v] = sigmoidf_(az[v]); rf[nt][v] = sigmoidf_(ar[v]); }
  }
  __syncthreads();

  // Phase B weights; overwrite A cols 0..63 with bf16(r*h) (aggr half untouched)
  stage_w_frag<128, 64>(Wh, sWa, tid, 64);
  stage_w_frag<128, 64>(Wd, sWb, tid, 64);
#pragma unroll
  for (int nt = 0; nt < 4; ++nt)
#pragma unroll
    for (int v = 0; v < 8; ++v) {
      int m2 = v + hf * 8, c = n0 + nt * 16;
      sA[wv][afrag_idx(m2, c)] = f2bf(rf[nt][v] * sH[wv][m2 * 64 + c]);
    }
  __syncthreads();

  // hc = relu([r*h|aggr] @ Wh + bh); hnew = (1-z)h + z*hc   (16 WMMA)
  v8f hn[4];
#pragma unroll
  for (int nt = 0; nt < 4; ++nt) {
    v8f ah;
    float bv = sB[128 + n0 + nt * 16];
#pragma unroll
    for (int v = 0; v < 8; ++v) ah[v] = bv;
#pragma unroll
    for (int ks = 0; ks < 4; ++ks) {
      v16bf a = lds_load_frag(afrag_ptr(sA[wv], ks, lane));
      v16bf b = lds_load_frag(bfrag_ptr(sWa, 4, ks, nt, lane));
      ah = wmma_bf16(a, b, ah);
    }
#pragma unroll
    for (int v = 0; v < 8; ++v) {
      float hc = reluf(ah[v]);
      float z  = zf[nt][v];
      float ho = sH[wv][(v + hf * 8) * 64 + n0 + nt * 16];
      hn[nt][v] = (1.f - z) * ho + z * hc;
    }
  }
  __syncthreads();

  // A = [node_id | hnew]
  const float4* Ir = (const float4*)(node_id + (size_t)node * 64);
#pragma unroll
  for (int q = 0; q < 8; ++q) {
    int f = hf * 8 + q, c = f * 4;
    store_bf4_frag(sA[wv], m, c, Ir[f]);
  }
#pragma unroll
  for (int nt = 0; nt < 4; ++nt)
#pragma unroll
    for (int v = 0; v < 8; ++v)
      sA[wv][afrag_idx(v + hf * 8, 64 + n0 + nt * 16)] = f2bf(hn[nt][v]);
  __syncthreads();

  // h = relu(A @ Wd + bd)   (16 WMMA) -> global (in-place update is safe)
#pragma unroll
  for (int nt = 0; nt < 4; ++nt) {
    v8f ad;
    float bv = sB[192 + n0 + nt * 16];
#pragma unroll
    for (int v = 0; v < 8; ++v) ad[v] = bv;
#pragma unroll
    for (int ks = 0; ks < 4; ++ks) {
      v16bf a = lds_load_frag(afrag_ptr(sA[wv], ks, lane));
      v16bf b = lds_load_frag(bfrag_ptr(sWb, 4, ks, nt, lane));
      ad = wmma_bf16(a, b, ad);
    }
#pragma unroll
    for (int v = 0; v < 8; ++v) {
      int node2 = base + v + hf * 8;
      if (node2 < N) h[(size_t)node2 * 64 + n0 + nt * 16] = reluf(ad[v]);
    }
  }
}

// ---------------------------------------------------------------------------
// Head: xo = relu(LN([node_id|h]@Wf1 + bf1)); out = xo@Wf2 + bf2
// ---------------------------------------------------------------------------
#define HK_WAVES 4
__global__ __launch_bounds__(128) void head_kernel(
    const float* __restrict__ node_id, const float* __restrict__ h,
    const float* __restrict__ Wf1, const float* __restrict__ bf1,
    const float* __restrict__ lng, const float* __restrict__ lnb,
    const float* __restrict__ Wf2, const float* __restrict__ bf2,
    float* __restrict__ out, int N) {
  __shared__ alignas(32) unsigned short sWp[128 * 64];
  __shared__ float          sB[64 * 3 + 128 + 2];
  __shared__ alignas(32) unsigned short sA[HK_WAVES][16 * 128];
  __shared__ float          sO[HK_WAVES][16 * 64];

  const int tid = threadIdx.x, lane = tid & 31, wv = tid >> 5;
  stage_w_frag<128, 64>(Wf1, sWp, tid, 128);
  if (tid < 64)  { sB[tid] = bf1[tid]; sB[64 + tid] = lng[tid]; sB[128 + tid] = lnb[tid]; }
  if (tid < 128) sB[192 + tid] = Wf2[tid];
  if (tid < 2)   sB[320 + tid] = bf2[tid];

  const int base = (blockIdx.x * HK_WAVES + wv) * 16;
  const int m = lane & 15, hf = lane >> 4, n0 = lane & 15;
  int node = base + m; if (node >= N) node = N - 1;
  const float4* Ir = (const float4*)(node_id + (size_t)node * 64);
  const float4* Hr = (const float4*)(h + (size_t)node * 64);
#pragma unroll
  for (int q = 0; q < 8; ++q) {
    int f = hf * 8 + q, c = f * 4;
    store_bf4_frag(sA[wv], m, c, Ir[f]);
    store_bf4_frag(sA[wv], m, 64 + c, Hr[f]);
  }
  __syncthreads();

#pragma unroll
  for (int nt = 0; nt < 4; ++nt) {
    v8f acc;
    float bv = sB[n0 + nt * 16];
#pragma unroll
    for (int v = 0; v < 8; ++v) acc[v] = bv;
#pragma unroll
    for (int ks = 0; ks < 4; ++ks) {
      v16bf a = lds_load_frag(afrag_ptr(sA[wv], ks, lane));
      v16bf b = lds_load_frag(bfrag_ptr(sWp, 4, ks, nt, lane));
      acc = wmma_bf16(a, b, acc);
    }
#pragma unroll
    for (int v = 0; v < 8; ++v) sO[wv][(v + hf * 8) * 64 + n0 + nt * 16] = acc[v];
  }
  __syncthreads();

  if (lane < 16) {
    int nd = base + lane;
    if (nd < N) {
      const float* row = &sO[wv][lane * 64];
      float mean = 0.f;
#pragma unroll
      for (int c = 0; c < 64; ++c) mean += row[c];
      mean *= (1.f / 64.f);
      float var = 0.f;
#pragma unroll
      for (int c = 0; c < 64; ++c) { float d = row[c] - mean; var += d * d; }
      var *= (1.f / 64.f);
      float rs = rsqrtf(var + LN_EPS);
      float o0 = sB[320], o1 = sB[321];
#pragma unroll
      for (int c = 0; c < 64; ++c) {
        float y = reluf((row[c] - mean) * rs * sB[64 + c] + sB[128 + c]);
        o0 += y * sB[192 + c * 2 + 0];
        o1 += y * sB[192 + c * 2 + 1];
      }
      out[nd * 2 + 0] = o0;
      out[nd * 2 + 1] = o1;
    }
  }
}

// ---------------------------------------------------------------------------
// Launch
// ---------------------------------------------------------------------------
extern "C" void kernel_launch(void* const* d_in, const int* in_sizes, int n_in,
                              void* d_out, int out_size, void* d_ws, size_t ws_size,
                              hipStream_t stream) {
  (void)n_in; (void)out_size; (void)ws_size;
  const float*     x  = (const float*)d_in[0];
  const long long* ei = (const long long*)d_in[1];     // int64 edge_index (2,E)
  const int N    = in_sizes[0] / 16;
  const int E    = in_sizes[1] / 2;
  const int Etot = E + N;

  const float* W_pre = (const float*)d_in[2];
  const float* b_pre = (const float*)d_in[3];
  const float* lnpg  = (const float*)d_in[4];
  const float* lnpb  = (const float*)d_in[5];
  const float* W_id  = (const float*)d_in[6];
  const float* b_id  = (const float*)d_in[7];
  const float* W_h0  = (const float*)d_in[8];
  const float* b_h0  = (const float*)d_in[9];
  const float* Wg0   = (const float*)d_in[10];
  const float* bg0   = (const float*)d_in[11];
  const float* Wg1   = (const float*)d_in[12];
  const float* bg1   = (const float*)d_in[13];
  const float* Wgo   = (const float*)d_in[14];
  const float* bgo   = (const float*)d_in[15];
  const float* Wz    = (const float*)d_in[16];
  const float* bz    = (const float*)d_in[17];
  const float* Wr    = (const float*)d_in[18];
  const float* br    = (const float*)d_in[19];
  const float* Wh    = (const float*)d_in[20];
  const float* bh    = (const float*)d_in[21];
  const float* Wd    = (const float*)d_in[22];
  const float* bd    = (const float*)d_in[23];
  const float* Wf1   = (const float*)d_in[24];
  const float* bf1   = (const float*)d_in[25];
  const float* lnf1g = (const float*)d_in[26];
  const float* lnf1b = (const float*)d_in[27];
  const float* Wf2   = (const float*)d_in[28];
  const float* bf2   = (const float*)d_in[29];

  char* ws = (char*)d_ws;
  auto take = [&](size_t bytes) { char* p = ws; ws += (bytes + 255) & ~(size_t)255; return p; };
  float* dis  = (float*)take((size_t)N * 4);          // deg -> deg^-0.5 in place
  float* nrm  = (float*)take((size_t)Etot * 4);
  float* nid  = (float*)take((size_t)N * 64 * 4);
  float* hbuf = (float*)take((size_t)N * 64 * 4);
  float* aggr = (float*)take((size_t)N * 64 * 4);

  // degree (self loop folded into init = 1) -> rsqrt -> per-edge norm
  fill_kernel<<<(N + 255) / 256, 256, 0, stream>>>(dis, 1.0f, N);
  deg_accum_kernel<<<(E + 255) / 256, 256, 0, stream>>>(ei + E, E, dis);
  rsqrt_kernel<<<(N + 255) / 256, 256, 0, stream>>>(dis, N);
  norm_kernel<<<(Etot + 255) / 256, 256, 0, stream>>>(ei, ei + E, E, Etot, dis, nrm);

  preproc_kernel<<<(N + 255) / 256, 256, 0, stream>>>(x, W_pre, b_pre, lnpg, lnpb,
                                                      W_id, b_id, W_h0, b_h0, nid, hbuf, N);

  const int etiles  = (Etot + 15) / 16;
  const int eblocks = (etiles + EK_WAVES - 1) / EK_WAVES;
  const int ntiles  = (N + 15) / 16;
  const int gblocks = (ntiles + GK_WAVES - 1) / GK_WAVES;
  const int zfill   = ((N * 64) + 255) / 256;

  for (int layer = 0; layer < 3; ++layer) {
    fill_kernel<<<zfill, 256, 0, stream>>>(aggr, 0.0f, N * 64);
    edge_gate_kernel<<<eblocks, 128, 0, stream>>>(hbuf, nrm, ei, ei + E, E, Etot,
                                                  Wg0, bg0, Wg1, bg1, Wgo, bgo, aggr);
    gru_kernel<<<gblocks, 64, 0, stream>>>(hbuf, aggr, nid,
                                           Wz, bz, Wr, br, Wh, bh, Wd, bd, N);
  }

  const int hblocks = (ntiles + HK_WAVES - 1) / HK_WAVES;
  head_kernel<<<hblocks, 128, 0, stream>>>(nid, hbuf, Wf1, bf1, lnf1g, lnf1b,
                                           Wf2, bf2, (float*)d_out, N);
}
